// GNN_44057774522952
// MI455X (gfx1250) — matrix operand order repaired and checked by reading, not compiled, not run on hardware
//
#include <hip/hip_runtime.h>
#include <math.h>

#define HC 512
#define EDIM 16
#define HEADS 4
#define EMB 128
#define OUTD 64
#define NEG_SLOPE 0.2f

typedef __attribute__((ext_vector_type(16))) __bf16 v16bf;
typedef __attribute__((ext_vector_type(8)))  float  v8f;
typedef __attribute__((ext_vector_type(8)))  unsigned short ushort8;

union FragU { v16bf v; ushort8 h[2]; unsigned short u[16]; };
union BfU { __bf16 b; unsigned short u; };

__device__ __forceinline__ unsigned short f2bf(float f) {
    BfU t; t.b = (__bf16)f;            // native v_cvt_*_bf16_f32 (RNE)
    return t.u;
}

// ---------------------------------------------------------------------------
// Weight prep: W[K][Nc] f32  ->  Bt[Nc][K] bf16 (transposed, fragment-friendly)
// ---------------------------------------------------------------------------
__global__ void wt_to_bf16t(const float* __restrict__ W, unsigned short* __restrict__ Bt,
                            int K, int Nc)
{
    long long i = (long long)blockIdx.x * blockDim.x + threadIdx.x;
    if (i >= (long long)K * Nc) return;
    int n = (int)(i / K);
    int k = (int)(i - (long long)n * K);
    Bt[i] = f2bf(W[(size_t)k * Nc + n]);
}

// ---------------------------------------------------------------------------
// Tiled WMMA GEMM: C[M x Nc] = A[M x K] * Bt^T + bias, optional relu.
// Block tile 128x64, 8 waves (4 along M x 2 along N), each wave 32x32 via
// 2x2 v_wmma_f32_16x16x32_bf16. LDS tiles are K-contiguous per row so every
// fragment is two 16B ds_load_b128's (stride 40 ushorts = 80B: 16B-aligned,
// 5 coprime 16 -> conflict-free). B tile streams via async global->LDS DMA.
// ---------------------------------------------------------------------------
#define BM 128
#define BN 64
#define BK 32
#define LDA 40
#define LDBT 40

__global__ __launch_bounds__(256)
void gemm_bias_kernel(const float* __restrict__ A, const unsigned short* __restrict__ Bt,
                      const float* __restrict__ bias, float* __restrict__ C,
                      int M, int K, int Nc, int relu)
{
    __shared__ unsigned short sA[BM * LDA];   // 10 KB
    __shared__ unsigned short sBt[BN * LDBT]; // 5 KB
    const int tid  = threadIdx.x;
    const int lane = tid & 31;
    const int wave = tid >> 5;
    const int wm0  = (wave & 3) * 32;
    const int wn0  = (wave >> 2) * 32;
    const int m0g  = blockIdx.x * BM;
    const int n0g  = blockIdx.y * BN;

    v8f zero = {0.f,0.f,0.f,0.f,0.f,0.f,0.f,0.f};
    v8f acc[2][2];
    acc[0][0] = zero; acc[0][1] = zero; acc[1][0] = zero; acc[1][1] = zero;

    const int rsel = lane & 15;
    const int kb1  = (lane < 16) ? 0 : 8;    // A-frag K-phase (ISA 7.12.2)
    const int kbB  = (lane < 16) ? 0 : 8;    // B-frag K-phase

    // fill-stage index precompute
    const int ar  = tid >> 3;          // A: row within tile (0..31 per pass)
    const int ac4 = (tid & 7) * 4;     // A: 4-float chunk within BK
    const int bn_ = tid >> 2;          // B: row n within tile (0..63)
    const int bk8 = (tid & 3) * 8;     // B: 8-elem chunk within BK

    // per-thread LDS destination (wave-relative offset = low 32 bits of generic addr)
    const unsigned ldsB = (unsigned)(unsigned long long)&sBt[bn_ * LDBT + bk8];

    for (int kb = 0; kb < K; kb += BK) {
        // ---- B tile: async global -> LDS (ASYNCcnt), 16B per lane ----
        {
            const unsigned short* src = &Bt[(size_t)(n0g + bn_) * K + kb + bk8];
            asm volatile("global_load_async_to_lds_b128 %0, %1, off"
                         :: "v"(ldsB), "v"(src) : "memory");
        }
        // ---- A tile: BM x BK, float4 loads -> packed bf16 ds_store_b64 ----
        #pragma unroll
        for (int it = 0; it < 4; ++it) {
            int r  = ar + it * 32;
            int gr = m0g + r;
            float4 f = {0.f, 0.f, 0.f, 0.f};
            if (gr < M) {
                f = *(const float4*)&A[(size_t)gr * K + kb + ac4];
                // speculative prefetch of next K chunk (OOB is silently dropped)
                __builtin_prefetch(&A[(size_t)gr * K + kb + BK + ac4], 0, 1);
            }
            unsigned p0 = (unsigned)f2bf(f.x) | ((unsigned)f2bf(f.y) << 16);
            unsigned p1 = (unsigned)f2bf(f.z) | ((unsigned)f2bf(f.w) << 16);
            uint2 pk; pk.x = p0; pk.y = p1;
            *(uint2*)&sA[r * LDA + ac4] = pk;
        }
        asm volatile("s_wait_asynccnt 0" ::: "memory");
        __syncthreads();

        FragU a[2], b[2];
        #pragma unroll
        for (int am = 0; am < 2; ++am) {
            int mrow = wm0 + am * 16 + rsel;
            a[am].h[0] = *(const ushort8*)&sA[mrow * LDA + kb1];       // k = kb1..kb1+7
            a[am].h[1] = *(const ushort8*)&sA[mrow * LDA + kb1 + 16];  // k = kb1+16..+23
        }
        #pragma unroll
        for (int bn = 0; bn < 2; ++bn) {
            int ncol = wn0 + bn * 16 + rsel;
            b[bn].h[0] = *(const ushort8*)&sBt[ncol * LDBT + kbB];     // k = kbB..kbB+7
            b[bn].h[1] = *(const ushort8*)&sBt[ncol * LDBT + kbB + 16]; // k = kbB+16..+23
        }
        #pragma unroll
        for (int am = 0; am < 2; ++am)
            #pragma unroll
            for (int bn = 0; bn < 2; ++bn)
                acc[am][bn] = __builtin_amdgcn_wmma_f32_16x16x32_bf16(
                    false, a[am].v, false, b[bn].v, (short)0, acc[am][bn], false, false);
        __syncthreads();
    }

    #pragma unroll
    for (int am = 0; am < 2; ++am)
        #pragma unroll
        for (int bn = 0; bn < 2; ++bn) {
            int colg  = n0g + wn0 + bn * 16 + rsel;
            int rbase = m0g + wm0 + am * 16 + ((lane >= 16) ? 8 : 0);
            float bv = bias[colg];
            #pragma unroll
            for (int i = 0; i < 8; ++i) {
                int rg = rbase + i;
                if (rg < M) {
                    float v = acc[am][bn][i] + bv;
                    if (relu) v = fmaxf(v, 0.f);
                    C[(size_t)rg * Nc + colg] = v;
                }
            }
        }
}

// ---------------------------------------------------------------------------
// Graph preprocessing: per-dst edge count and mean edge_attr for self loops.
// ---------------------------------------------------------------------------
__global__ void fill_f32(float* p, float v, long long n) {
    long long i = (long long)blockIdx.x * blockDim.x + threadIdx.x;
    if (i < n) p[i] = v;
}
__global__ void fill_u32(unsigned* p, unsigned v, long long n) {
    long long i = (long long)blockIdx.x * blockDim.x + threadIdx.x;
    if (i < n) p[i] = v;
}
__global__ void count_kernel(const int* __restrict__ ei, const float* __restrict__ ea,
                             float* cnt, float* lattr, int nE) {
    long long i = (long long)blockIdx.x * blockDim.x + threadIdx.x;
    if (i >= (long long)nE * EDIM) return;
    int e = (int)(i >> 4), k = (int)(i & 15);
    int d = ei[nE + e];
    atomicAdd(&lattr[(size_t)d * EDIM + k], ea[i]);
    if (k == 0) atomicAdd(&cnt[d], 1.f);
}
__global__ void loopnorm_kernel(const float* cnt, float* lattr, int nN) {
    long long i = (long long)blockIdx.x * blockDim.x + threadIdx.x;
    if (i >= (long long)nN * EDIM) return;
    lattr[i] /= fmaxf(cnt[i >> 4], 1.f);
}
__global__ void out_init_kernel(float* out, const float* bias, int nN) {
    long long i = (long long)blockIdx.x * blockDim.x + threadIdx.x;
    if (i >= (long long)nN * HC) return;
    out[i] = bias[i & (HC - 1)];
}

// ---------------------------------------------------------------------------
// Pass A: per-edge logits (ef computed on the fly from LDS-resident We),
// ordered-uint atomicMax for segment max. One wave per edge; lane group of 8
// per head, channel c = h*128 + j*8 + (lane&7).
// ---------------------------------------------------------------------------
__global__ __launch_bounds__(256)
void edge_logits_kernel(const float* __restrict__ xl, const float* __restrict__ xr,
                        const float* __restrict__ ea, const float* __restrict__ lattr,
                        const float* __restrict__ We, const float* __restrict__ att,
                        const int* __restrict__ ei, float* __restrict__ logit,
                        unsigned* __restrict__ mxi, int nE, int nN)
{
    __shared__ float sWe[EDIM * HC];   // 32 KB (LDS per WGP is 320 KB)
    for (int i = threadIdx.x; i < EDIM * HC; i += 256) sWe[i] = We[i];
    __syncthreads();

    const int lane = threadIdx.x & 31;
    const int wave = threadIdx.x >> 5;
    const int h = lane >> 3, p = lane & 7;

    float areg[16];
    #pragma unroll
    for (int j = 0; j < 16; ++j) areg[j] = att[h * EMB + j * 8 + p];

    const long long Etot = (long long)nE + nN;
    for (long long e = (long long)blockIdx.x * 8 + wave; e < Etot;
         e += (long long)gridDim.x * 8) {
        int s, d; const float* eap;
        if (e < nE) { s = ei[e]; d = ei[nE + e]; eap = ea + e * EDIM; }
        else        { s = d = (int)(e - nE);     eap = lattr + (size_t)s * EDIM; }
        float ev[16];
        #pragma unroll
        for (int k = 0; k < 16; ++k) ev[k] = eap[k];
        const float* xlp = xl + (size_t)s * HC;
        const float* xrp = xr + (size_t)d * HC;
        float accl = 0.f;
        #pragma unroll
        for (int j = 0; j < 16; ++j) {
            int c = h * EMB + j * 8 + p;
            float ef = 0.f;
            #pragma unroll
            for (int k = 0; k < 16; ++k) ef += ev[k] * sWe[k * HC + c];
            float v = xlp[c] + xrp[c] + ef;
            v = (v > 0.f) ? v : NEG_SLOPE * v;
            accl += areg[j] * v;
        }
        accl += __shfl_xor(accl, 1, 32);
        accl += __shfl_xor(accl, 2, 32);
        accl += __shfl_xor(accl, 4, 32);
        if (p == 0) {
            logit[e * HEADS + h] = accl;
            unsigned bits = __float_as_uint(accl);
            unsigned key = bits ^ (((int)bits < 0) ? 0xFFFFFFFFu : 0x80000000u);
            atomicMax(&mxi[(size_t)d * HEADS + h], key);
        }
    }
}

// Pass B: num = exp(logit - max); accumulate denominator.
__global__ void softmax_den_kernel(const int* __restrict__ ei, float* __restrict__ logit,
                                   const unsigned* __restrict__ mxi, float* __restrict__ den,
                                   int nE, int nN)
{
    long long i = (long long)blockIdx.x * blockDim.x + threadIdx.x;
    long long tot = (long long)(nE + nN) * HEADS;
    if (i >= tot) return;
    int e = (int)(i >> 2), h = (int)(i & 3);
    int d = (e < nE) ? ei[nE + e] : (e - nE);
    unsigned key = mxi[(size_t)d * HEADS + h];
    unsigned bits = (key & 0x80000000u) ? (key ^ 0x80000000u) : ~key;
    float num = expf(logit[i] - __uint_as_float(bits));
    logit[i] = num;
    atomicAdd(&den[(size_t)d * HEADS + h], num);
}

// Pass C: out[dst] += alpha * xl[src]  (one wave per edge, 16 ch per lane).
__global__ __launch_bounds__(256)
void scatter_out_kernel(const float* __restrict__ xl, const float* __restrict__ logit,
                        const float* __restrict__ den, const int* __restrict__ ei,
                        float* __restrict__ out, int nE, int nN)
{
    const int lane = threadIdx.x & 31;
    const int wave = threadIdx.x >> 5;
    const int h = lane >> 3, p = lane & 7;
    const long long Etot = (long long)nE + nN;
    for (long long e = (long long)blockIdx.x * 8 + wave; e < Etot;
         e += (long long)gridDim.x * 8) {
        int s, d;
        if (e < nE) { s = ei[e]; d = ei[nE + e]; }
        else        { s = d = (int)(e - nE); }
        float alpha = logit[e * HEADS + h] / (den[(size_t)d * HEADS + h] + 1e-16f);
        const float* xlp = xl + (size_t)s * HC;
        float* op = out + (size_t)d * HC;
        #pragma unroll
        for (int j = 0; j < 16; ++j) {
            int c = h * EMB + j * 8 + p;
            atomicAdd(&op[c], alpha * xlp[c]);
        }
    }
}

static inline int cdiv_ll(long long a, long long b) { return (int)((a + b - 1) / b); }

extern "C" void kernel_launch(void* const* d_in, const int* in_sizes, int n_in,
                              void* d_out, int out_size, void* d_ws, size_t ws_size,
                              hipStream_t stream)
{
    const float* x     = (const float*)d_in[0];
    const int*   ei    = (const int*)  d_in[1];
    const float* ea    = (const float*)d_in[2];
    const float* Wl1   = (const float*)d_in[3];
    const float* bl1   = (const float*)d_in[4];
    const float* Wr1   = (const float*)d_in[5];
    const float* br1   = (const float*)d_in[6];
    const float* We1   = (const float*)d_in[7];
    const float* att1  = (const float*)d_in[8];
    const float* bias1 = (const float*)d_in[9];
    const float* Wl2   = (const float*)d_in[10];
    const float* bl2   = (const float*)d_in[11];
    const float* Wr2   = (const float*)d_in[12];
    const float* br2   = (const float*)d_in[13];
    const float* We2   = (const float*)d_in[14];
    const float* att2  = (const float*)d_in[15];
    const float* bias2 = (const float*)d_in[16];
    const float* Wd1   = (const float*)d_in[17];
    const float* bd1   = (const float*)d_in[18];
    const float* Wd2   = (const float*)d_in[19];
    const float* bd2   = (const float*)d_in[20];

    const int nN = in_sizes[0] / 128;   // NODE_DIM
    const int nE = in_sizes[2] / EDIM;
    const long long Etot = (long long)nE + nN;

    // workspace layout (f32): 3 node buffers + logits + graph stats + bf16 weights
    float* buf0  = (float*)d_ws;
    float* buf1  = buf0 + (size_t)nN * HC;
    float* buf2  = buf1 + (size_t)nN * HC;
    float* logit = buf2 + (size_t)nN * HC;
    float* cnt   = logit + (size_t)Etot * HEADS;
    float* lattr = cnt + nN;
    float* den   = lattr + (size_t)nN * EDIM;
    unsigned* mxi = (unsigned*)(den + (size_t)nN * HEADS);
    unsigned short* btbuf = (unsigned short*)(mxi + (size_t)nN * HEADS); // 512KB max

    const unsigned ENC_NEG_INF = 0x007FFFFFu; // ordered-uint encoding of -inf

    // --- graph preprocessing (self-loop mean edge_attr) ---
    fill_f32<<<cdiv_ll(nN, 256), 256, 0, stream>>>(cnt, 0.f, nN);
    fill_f32<<<cdiv_ll((long long)nN * EDIM, 256), 256, 0, stream>>>(lattr, 0.f, (long long)nN * EDIM);
    count_kernel<<<cdiv_ll((long long)nE * EDIM, 256), 256, 0, stream>>>(ei, ea, cnt, lattr, nE);
    loopnorm_kernel<<<cdiv_ll((long long)nN * EDIM, 256), 256, 0, stream>>>(cnt, lattr, nN);

    dim3 gHC(cdiv_ll(nN, BM), HC / BN);
    const int cw128 = cdiv_ll((long long)128 * HC, 256);
    const int cw512 = cdiv_ll((long long)HC * HC, 256);

    // --- layer 1 ---
    wt_to_bf16t<<<cw128, 256, 0, stream>>>(Wl1, btbuf, 128, HC);
    gemm_bias_kernel<<<gHC, 256, 0, stream>>>(x, btbuf, bl1, buf0, nN, 128, HC, 0);
    wt_to_bf16t<<<cw128, 256, 0, stream>>>(Wr1, btbuf, 128, HC);
    gemm_bias_kernel<<<gHC, 256, 0, stream>>>(x, btbuf, br1, buf1, nN, 128, HC, 0);
    fill_f32<<<cdiv_ll((long long)nN * HEADS, 256), 256, 0, stream>>>(den, 0.f, (long long)nN * HEADS);
    fill_u32<<<cdiv_ll((long long)nN * HEADS, 256), 256, 0, stream>>>(mxi, ENC_NEG_INF, (long long)nN * HEADS);
    out_init_kernel<<<cdiv_ll((long long)nN * HC, 256), 256, 0, stream>>>(buf2, bias1, nN);
    edge_logits_kernel<<<4096, 256, 0, stream>>>(buf0, buf1, ea, lattr, We1, att1, ei, logit, mxi, nE, nN);
    softmax_den_kernel<<<cdiv_ll(Etot * HEADS, 256), 256, 0, stream>>>(ei, logit, mxi, den, nE, nN);
    scatter_out_kernel<<<4096, 256, 0, stream>>>(buf0, logit, den, ei, buf2, nE, nN);

    // --- layer 2 (h1 in buf2) ---
    wt_to_bf16t<<<cw512, 256, 0, stream>>>(Wl2, btbuf, HC, HC);
    gemm_bias_kernel<<<gHC, 256, 0, stream>>>(buf2, btbuf, bl2, buf0, nN, HC, HC, 0);
    wt_to_bf16t<<<cw512, 256, 0, stream>>>(Wr2, btbuf, HC, HC);
    gemm_bias_kernel<<<gHC, 256, 0, stream>>>(buf2, btbuf, br2, buf1, nN, HC, HC, 0);
    fill_f32<<<cdiv_ll((long long)nN * HEADS, 256), 256, 0, stream>>>(den, 0.f, (long long)nN * HEADS);
    fill_u32<<<cdiv_ll((long long)nN * HEADS, 256), 256, 0, stream>>>(mxi, ENC_NEG_INF, (long long)nN * HEADS);
    out_init_kernel<<<cdiv_ll((long long)nN * HC, 256), 256, 0, stream>>>(buf2, bias2, nN);
    edge_logits_kernel<<<4096, 256, 0, stream>>>(buf0, buf1, ea, lattr, We2, att2, ei, logit, mxi, nE, nN);
    softmax_den_kernel<<<cdiv_ll(Etot * HEADS, 256), 256, 0, stream>>>(ei, logit, mxi, den, nE, nN);
    scatter_out_kernel<<<4096, 256, 0, stream>>>(buf0, logit, den, ei, buf2, nE, nN);

    // --- decoder MLP (h2 in buf2) ---
    wt_to_bf16t<<<cw512, 256, 0, stream>>>(Wd1, btbuf, HC, HC);
    gemm_bias_kernel<<<gHC, 256, 0, stream>>>(buf2, btbuf, bd1, buf0, nN, HC, HC, 1);
    wt_to_bf16t<<<cdiv_ll((long long)HC * OUTD, 256), 256, 0, stream>>>(Wd2, btbuf, HC, OUTD);
    dim3 gOut(cdiv_ll(nN, BM), OUTD / BN);
    gemm_bias_kernel<<<gOut, 256, 0, stream>>>(buf0, btbuf, bd2, (float*)d_out, nN, HC, OUTD, 0);

    (void)n_in; (void)out_size; (void)ws_size;
}
